// SelectiveMultiHeadAttention_16647293239359
// MI455X (gfx1250) — matrix operand order repaired and checked by reading, not compiled, not run on hardware
//
#include <hip/hip_runtime.h>

#define HIDDEN   1024
#define HEAD     64
#define NHEADS   16
#define BATCH    2
#define SEQ      2048
#define AH       (HEAD * NHEADS)   // 1024

typedef _Float16 half8  __attribute__((ext_vector_type(8)));
typedef _Float16 half16 __attribute__((ext_vector_type(16)));
typedef float    f32x8  __attribute__((ext_vector_type(8)));
typedef unsigned u32x4  __attribute__((ext_vector_type(4)));
typedef int      i32x4v __attribute__((ext_vector_type(4)));
typedef int      i32x8v __attribute__((ext_vector_type(8)));

// Tensor Data Mover availability (toolchain-dependent arity, see probe notes)
#if defined(__has_builtin)
#  if __has_builtin(__builtin_amdgcn_tensor_load_to_lds) && \
      __has_builtin(__builtin_amdgcn_s_wait_tensorcnt)
#    define HAVE_TDM 1
#  endif
#endif
#ifndef HAVE_TDM
#  define HAVE_TDM 0
#endif

// ---------------------------------------------------------------------------
// Fragment loaders for v_wmma_f32_16x16x32_f16 (wave32).
// A (16x32, MxK): lane L: M=L%16, hi=L/16;
//   halves j -> K = hi*16 + j
// B (32x16, KxN): lane L: N=L%16, hi=L/16; halves j -> K = hi*16 + j
// C/D (16x16 f32): elem r in lane L -> row = r + 8*(L/16), col = L%16
// ---------------------------------------------------------------------------
__device__ __forceinline__ half16 load_a_frag(const _Float16* p, int ld) {
  const int lane = threadIdx.x & 31;
  const int m    = lane & 15;
  const int hi   = lane >> 4;
  const _Float16* r0 = p + (size_t)m * ld + hi * 16;
  half8 lo = *(const half8*)(r0);
  half8 hv = *(const half8*)(r0 + 8);
  half16 out;
#pragma unroll
  for (int i = 0; i < 8; ++i) { out[i] = lo[i]; out[i + 8] = hv[i]; }
  return out;
}

__device__ __forceinline__ half16 load_b_frag(const _Float16* p, int ld) {
  const int lane = threadIdx.x & 31;
  const int n    = lane & 15;
  const int hi   = lane >> 4;
  return *(const half16*)(p + (size_t)n * ld + hi * 16);
}

// ---------------------------------------------------------------------------
// 16-lane-row reductions via DPP row_ror rotate-reduce (pure VALU, co-executes
// with WMMA; DPP row = 16 lanes = the C-fragment half-wave row group).
// ---------------------------------------------------------------------------
#if defined(__has_builtin) && __has_builtin(__builtin_amdgcn_update_dpp)
template <int CTRL>
__device__ __forceinline__ float dpp_rot(float x) {
  return __int_as_float(
      __builtin_amdgcn_update_dpp(0, __float_as_int(x), CTRL, 0xf, 0xf, true));
}
__device__ __forceinline__ float rowmax16(float x) {
  x = fmaxf(x, dpp_rot<0x121>(x));
  x = fmaxf(x, dpp_rot<0x122>(x));
  x = fmaxf(x, dpp_rot<0x124>(x));
  x = fmaxf(x, dpp_rot<0x128>(x));
  return x;
}
__device__ __forceinline__ float rowsum16(float x) {
  x += dpp_rot<0x121>(x);
  x += dpp_rot<0x122>(x);
  x += dpp_rot<0x124>(x);
  x += dpp_rot<0x128>(x);
  return x;
}
#else
__device__ __forceinline__ float rowmax16(float x) {
#pragma unroll
  for (int msk = 1; msk <= 8; msk <<= 1) x = fmaxf(x, __shfl_xor(x, msk, 32));
  return x;
}
__device__ __forceinline__ float rowsum16(float x) {
#pragma unroll
  for (int msk = 1; msk <= 8; msk <<= 1) x += __shfl_xor(x, msk, 32);
  return x;
}
#endif

// ---------------------------------------------------------------------------
// TDM: issue a 2-D tensor tile load Global->LDS.
// 64 x 64 f16 tile; hardware row padding of 8 DWORDs every 32 DWORDs
// (pad_interval code 4, pad_amount code 7) reproduces the 80-half LDS pitch.
// D# per ISA 8.3/8.4: group0 = {flags, lds_addr, global_addr, type=2},
// group1 packs data_size/pad/tensor dims/tile dims/stride.
// ---------------------------------------------------------------------------
#if HAVE_TDM
__device__ __forceinline__ void tdm_issue_2d(const void* gsrc, unsigned lds_bytes,
                                             unsigned stride_elems) {
  const unsigned long long ga = (unsigned long long)(uintptr_t)gsrc;
  u32x4 g0;
  g0.x = 1u;                                                  // count=1, user D#
  g0.y = lds_bytes;                                           // lds_addr
  g0.z = (unsigned)ga;                                        // global_addr[31:0]
  g0.w = (unsigned)((ga >> 32) & 0x01FFFFFFu) | (2u << 30);   // [56:32] | type=2

  const unsigned td0   = stride_elems;   // tensor_dim0 (elements)
  const unsigned td1   = 1u << 22;       // tensor_dim1: large (always in-bounds)
  const unsigned tile0 = 64, tile1 = 64;
  i32x8v g1;
  g1[0] = (int)((1u << 16)     // data_size = 2 bytes
              | (1u << 20)     // pad_enable
              | (4u << 22)     // pad_interval: every 32 DWORDs
              | (7u << 25));   // pad_amount: 8 DWORDs
  g1[1] = (int)((td0 & 0xFFFFu) << 16);                   // abar=0 | td0[15:0]
  g1[2] = (int)((td0 >> 16) | ((td1 & 0xFFFFu) << 16));   // td0[31:16] | td1[15:0]
  g1[3] = (int)((td1 >> 16) | (tile0 << 16));             // td1[31:16] | tile_dim0
  g1[4] = (int)tile1;                                     // tile_dim1 | tile_dim2=0
  g1[5] = (int)td0;                                       // tensor_dim0_stride[31:0]
  g1[6] = 0;
  g1[7] = 0;
  const i32x4v z4 = {0, 0, 0, 0};
#if __clang_major__ >= 23
  const i32x8v z8 = {0, 0, 0, 0, 0, 0, 0, 0};
  __builtin_amdgcn_tensor_load_to_lds(g0, g1, z4, z4, z8, 0);
#else
  __builtin_amdgcn_tensor_load_to_lds(g0, g1, z4, z4, 0);
#endif
}
#endif

// ---------------------------------------------------------------------------
// f32 -> f16 staging
// ---------------------------------------------------------------------------
__global__ void __launch_bounds__(256)
cvt_f32_to_f16(const float* __restrict__ src, _Float16* __restrict__ dst, int n) {
  int i = blockIdx.x * blockDim.x + threadIdx.x;
  const int stride = gridDim.x * blockDim.x;
  for (; i < n; i += stride) dst[i] = (_Float16)src[i];
}

// ---------------------------------------------------------------------------
// GEMM:  out[M,N] = (A[M,K] * Bw[N,K]^T + bias) * scale
//   BIAS_ROW=false: bias[col] (y = x@W.T + b)
//   BIAS_ROW=true : bias[row] (used for v^T = Wv @ x^T + bv[row])
// Block = 256 threads = 8 waves arranged 2(M) x 4(N); wave tile = 16x64.
// ---------------------------------------------------------------------------
template <typename OutT, bool BIAS_ROW>
__global__ void __launch_bounds__(256)
gemm_bias_wmma(const _Float16* __restrict__ A,
               const _Float16* __restrict__ Bw,
               const float*    __restrict__ bias,
               OutT* __restrict__ out,
               int K, int N, float scale)
{
  const int wave  = threadIdx.x >> 5;
  const int lane  = threadIdx.x & 31;
  const int mBase = blockIdx.x * 32  + (wave >> 2) * 16;
  const int nBase = blockIdx.y * 256 + (wave & 3) * 64;

  f32x8 acc[4] = {};
  for (int k0 = 0; k0 < K; k0 += 32) {
    half16 a = load_a_frag(A + (size_t)mBase * K + k0, K);
#pragma unroll
    for (int nb = 0; nb < 4; ++nb) {
      half16 bf = load_b_frag(Bw + (size_t)(nBase + nb * 16) * K + k0, K);
      acc[nb] = __builtin_amdgcn_wmma_f32_16x16x32_f16(
          false, a, false, bf, (short)0, acc[nb], false, false);
    }
  }

  const int hi = lane >> 4, col0 = lane & 15;
#pragma unroll
  for (int nb = 0; nb < 4; ++nb) {
    const int col = nBase + nb * 16 + col0;
    const float bcol = BIAS_ROW ? 0.f : bias[col];
#pragma unroll
    for (int r = 0; r < 8; ++r) {
      const int row = mBase + r + 8 * hi;
      const float bv = BIAS_ROW ? bias[row] : bcol;
      out[(size_t)row * N + col] = (OutT)((acc[nb][r] + bv) * scale);
    }
  }
}

// ---------------------------------------------------------------------------
// Flash attention with double-buffered TDM tile pipeline.
// grid = (S/128, H, B), block = 256 (8 waves x 16 q-rows).
// Wave 0 issues TDM descriptors for tile i+1 while all waves compute tile i;
// TENSORcnt completes in-order, so s_wait_tensorcnt(2) = "current tile done,
// next tile still in flight". Fallback: synchronous coalesced b128 copies.
// ---------------------------------------------------------------------------
__global__ void __launch_bounds__(256)
flash_attn_wmma(const _Float16* __restrict__ q16,  // [B*S, AH]
                const _Float16* __restrict__ k16,  // [B*S, AH]
                const _Float16* __restrict__ vT,   // [AH, B*S]
                const float* __restrict__ bias,    // [S,S]
                const float* __restrict__ kappa,   // [H]
                _Float16* __restrict__ ctx)        // [B*S, AH]
{
  constexpr int KLD = 80, VLD = 80, PLD = 80;      // pad -> 32B-aligned rows
  __shared__ _Float16 Klds[2][64 * KLD];           // [buf][key][d]
  __shared__ _Float16 Vlds[2][64 * VLD];           // [buf][d][key]
  __shared__ _Float16 Plds[8 * 16 * PLD];          // wave-private P tiles

  const int b    = blockIdx.z;
  const int h    = blockIdx.y;
  const int q0   = blockIdx.x * 128;
  const int wave = threadIdx.x >> 5;
  const int lane = threadIdx.x & 31;
  const int hi   = lane >> 4, col0 = lane & 15;
  const int qRow0 = q0 + wave * 16;
  constexpr int NT = SEQ / 64;                     // 32 key tiles

  // Q fragments (pre-scaled by 64^-0.5 in the projection GEMM)
  const _Float16* qp = q16 + ((size_t)(b * SEQ + qRow0)) * AH + h * HEAD;
  const half16 qa0 = load_a_frag(qp, AH);
  const half16 qa1 = load_a_frag(qp + 32, AH);

  float m[8], l[8];
  f32x8 o[4] = {};
#pragma unroll
  for (int r = 0; r < 8; ++r) { m[r] = -1e30f; l[r] = 0.f; }

  _Float16* Pl = Plds + wave * 16 * PLD;

  // ---- tile staging (TDM or synchronous fallback) -------------------------
  auto stage = [&](int buf, int kb) {
#if HAVE_TDM
    if (wave == 0) {
      tdm_issue_2d(k16 + ((size_t)(b * SEQ + kb)) * AH + h * HEAD,
                   (unsigned)(uintptr_t)&Klds[buf][0], AH);
      tdm_issue_2d(vT + ((size_t)(h * HEAD)) * (BATCH * SEQ) + b * SEQ + kb,
                   (unsigned)(uintptr_t)&Vlds[buf][0], BATCH * SEQ);
    }
#else
    const int t    = threadIdx.x;
    const int rowk = t >> 2;            // key (K tile) / d (V tile)
    const int seg  = (t & 3) << 4;      // 0,16,32,48
    *(half16*)&Klds[buf][rowk * KLD + seg] =
        *(const half16*)(k16 + ((size_t)(b * SEQ + kb + rowk)) * AH + h * HEAD + seg);
    *(half16*)&Vlds[buf][rowk * VLD + seg] =
        *(const half16*)(vT + ((size_t)(h * HEAD + rowk)) * (BATCH * SEQ) + b * SEQ + kb + seg);
#endif
  };

  stage(0, 0);                                     // prologue: tile 0 in flight

  for (int it = 0; it < NT; ++it) {
    const int kb  = it * 64;
    const int cur = it & 1;

    if (it + 1 < NT) {
      stage((it + 1) & 1, kb + 64);                // overlap next tile's DMA
#if HAVE_TDM
      if (wave == 0) __builtin_amdgcn_s_wait_tensorcnt(2);
#endif
    } else {
#if HAVE_TDM
      if (wave == 0) __builtin_amdgcn_s_wait_tensorcnt(0);
#endif
    }
    __syncthreads();                               // publish buffer `cur`

    const _Float16* Kl = &Klds[cur][0];
    const _Float16* Vl = &Vlds[cur][0];

    // prefetch next bias tile rows into GL2 while we compute this block
    if (kb + 64 < SEQ)
      __builtin_prefetch(bias + (size_t)qRow0 * SEQ + kb + 64, 0, 1);

    // S = Q * K^T  (16x64 per wave, 8 WMMAs)
    f32x8 s[4] = {};
#pragma unroll
    for (int nb = 0; nb < 4; ++nb) {
      half16 b0 = load_b_frag(Kl + nb * 16 * KLD, KLD);
      s[nb] = __builtin_amdgcn_wmma_f32_16x16x32_f16(
          false, qa0, false, b0, (short)0, s[nb], false, false);
      half16 b1 = load_b_frag(Kl + nb * 16 * KLD + 32, KLD);
      s[nb] = __builtin_amdgcn_wmma_f32_16x16x32_f16(
          false, qa1, false, b1, (short)0, s[nb], false, false);
    }

    // + attention bias (broadcast over b,h)
#pragma unroll
    for (int nb = 0; nb < 4; ++nb) {
      const int kc = kb + nb * 16 + col0;
#pragma unroll
      for (int r = 0; r < 8; ++r)
        s[nb][r] += bias[(size_t)(qRow0 + r + 8 * hi) * SEQ + kc];
    }

    // online softmax (row stats replicated across each 16-lane half)
    float mnew[8], alpha[8];
#pragma unroll
    for (int r = 0; r < 8; ++r) {
      float mx = rowmax16(fmaxf(fmaxf(s[0][r], s[1][r]), fmaxf(s[2][r], s[3][r])));
      mnew[r]  = fmaxf(m[r], mx);
      alpha[r] = __expf(m[r] - mnew[r]);
      m[r]     = mnew[r];
    }
#pragma unroll
    for (int nb = 0; nb < 4; ++nb)
#pragma unroll
      for (int r = 0; r < 8; ++r)
        s[nb][r] = __expf(s[nb][r] - mnew[r]);
#pragma unroll
    for (int r = 0; r < 8; ++r) {
      float sm = rowsum16((s[0][r] + s[1][r]) + (s[2][r] + s[3][r]));
      l[r] = l[r] * alpha[r] + sm;
    }
#pragma unroll
    for (int db = 0; db < 4; ++db)
#pragma unroll
      for (int r = 0; r < 8; ++r)
        o[db][r] *= alpha[r];

    // P: C-layout -> A-layout via wave-private LDS tile (same-wave LDS ops
    // are in-order, no barrier needed)
#pragma unroll
    for (int nb = 0; nb < 4; ++nb)
#pragma unroll
      for (int r = 0; r < 8; ++r)
        Pl[(r + 8 * hi) * PLD + nb * 16 + col0] = (_Float16)s[nb][r];

    half16 pa0 = load_a_frag(Pl, PLD);
    half16 pa1 = load_a_frag(Pl + 32, PLD);

    // O += P * V  (8 WMMAs)
#pragma unroll
    for (int db = 0; db < 4; ++db) {
      half16 vb0 = load_b_frag(Vl + db * 16 * VLD, VLD);
      o[db] = __builtin_amdgcn_wmma_f32_16x16x32_f16(
          false, pa0, false, vb0, (short)0, o[db], false, false);
      half16 vb1 = load_b_frag(Vl + db * 16 * VLD + 32, VLD);
      o[db] = __builtin_amdgcn_wmma_f32_16x16x32_f16(
          false, pa1, false, vb1, (short)0, o[db], false, false);
    }

    __syncthreads();   // all waves done reading buffer `cur` -> reusable
  }

  // gate * O / l  -> context (f16)
  const float gate = 1.f / (1.f + __expf(-kappa[h]));
#pragma unroll
  for (int r = 0; r < 8; ++r) l[r] = gate / l[r];
#pragma unroll
  for (int db = 0; db < 4; ++db)
#pragma unroll
    for (int r = 0; r < 8; ++r) {
      const int qr = qRow0 + r + 8 * hi;
      ctx[((size_t)(b * SEQ + qr)) * AH + h * HEAD + db * 16 + col0] =
          (_Float16)(o[db][r] * l[r]);
    }
}

// ---------------------------------------------------------------------------
extern "C" void kernel_launch(void* const* d_in, const int* in_sizes, int n_in,
                              void* d_out, int out_size, void* d_ws, size_t ws_size,
                              hipStream_t stream) {
  (void)in_sizes; (void)n_in; (void)out_size; (void)ws_size;
  const float* query = (const float*)d_in[0];
  const float* abias = (const float*)d_in[1];
  const float* Wq    = (const float*)d_in[2];
  const float* bq    = (const float*)d_in[3];
  const float* Wk    = (const float*)d_in[4];
  const float* bk    = (const float*)d_in[5];
  const float* Wv    = (const float*)d_in[6];
  const float* bv    = (const float*)d_in[7];
  const float* Wo    = (const float*)d_in[8];
  const float* bo    = (const float*)d_in[9];
  const float* kappa = (const float*)d_in[10];
  float* out = (float*)d_out;

  const int M = BATCH * SEQ;                         // 4096
  _Float16* ws  = (_Float16*)d_ws;                   // ~48 MB f16 staging
  _Float16* x16  = ws; ws += (size_t)M * HIDDEN;
  _Float16* Wq16 = ws; ws += (size_t)AH * HIDDEN;
  _Float16* Wk16 = ws; ws += (size_t)AH * HIDDEN;
  _Float16* Wv16 = ws; ws += (size_t)AH * HIDDEN;
  _Float16* Wo16 = ws; ws += (size_t)HIDDEN * AH;
  _Float16* q16  = ws; ws += (size_t)M * AH;
  _Float16* k16  = ws; ws += (size_t)M * AH;
  _Float16* vT   = ws; ws += (size_t)AH * M;         // V stored transposed
  _Float16* c16  = ws;

  cvt_f32_to_f16<<<2048, 256, 0, stream>>>(query, x16, M * HIDDEN);
  cvt_f32_to_f16<<<1024, 256, 0, stream>>>(Wq, Wq16, AH * HIDDEN);
  cvt_f32_to_f16<<<1024, 256, 0, stream>>>(Wk, Wk16, AH * HIDDEN);
  cvt_f32_to_f16<<<1024, 256, 0, stream>>>(Wv, Wv16, AH * HIDDEN);
  cvt_f32_to_f16<<<1024, 256, 0, stream>>>(Wo, Wo16, HIDDEN * AH);

  // Q/K projections (Q pre-scaled by (HIDDEN/NUM_HEADS)^-0.5 = 0.125)
  dim3 gqkv(M / 32, AH / 256);
  gemm_bias_wmma<_Float16, false><<<gqkv, 256, 0, stream>>>(x16, Wq16, bq, q16, HIDDEN, AH, 0.125f);
  gemm_bias_wmma<_Float16, false><<<gqkv, 256, 0, stream>>>(x16, Wk16, bk, k16, HIDDEN, AH, 1.0f);

  // V projection with swapped operands: vT[AH, B*S] = Wv @ x^T + bv[row]
  dim3 gvt(AH / 32, M / 256);
  gemm_bias_wmma<_Float16, true><<<gvt, 256, 0, stream>>>(Wv16, x16, bv, vT, HIDDEN, M, 1.0f);

  // fused flash attention + head gating (TDM double-buffered tile pipeline)
  dim3 gatt(SEQ / 128, NHEADS, BATCH);
  flash_attn_wmma<<<gatt, 256, 0, stream>>>(q16, k16, vT, abias, kappa, c16);

  // output projection -> f32
  dim3 gout(M / 32, HIDDEN / 256);
  gemm_bias_wmma<float, false><<<gout, 256, 0, stream>>>(c16, Wo16, bo, out, AH, HIDDEN, 1.0f);
}
